// Joint_processing_unit_89756226552248
// MI455X (gfx1250) — matrix operand order repaired and checked by reading, not compiled, not run on hardware
//
#include <hip/hip_runtime.h>
#include <hip/hip_bf16.h>
#include <math.h>

typedef __attribute__((ext_vector_type(16))) _Float16 v16h;
typedef __attribute__((ext_vector_type(8)))  _Float16 v8h;
typedef __attribute__((ext_vector_type(8)))  float    v8f;
typedef __attribute__((ext_vector_type(4)))  float    f32x4;
typedef int i32x4 __attribute__((vector_size(16)));

#define B_   1024
#define LK_  1024
#define LQD_ 256
#define SQD_ 128

__device__ __forceinline__ float gelu_exact(float x) {
    return 0.5f * x * (1.0f + erff(x * 0.70710678118654752440f));
}

// ---- 16-lane XOR butterfly on the VALU (v_permlane16_b32), ds fallback ----
#if __has_builtin(__builtin_amdgcn_permlane16)
__device__ __forceinline__ float xor16p(float v, unsigned lo, unsigned hi) {
    unsigned iv = __float_as_uint(v);
    return __uint_as_float(__builtin_amdgcn_permlane16(iv, iv, lo, hi, false, false));
}
#define X16_1(v) xor16p((v), 0x67452301u, 0xEFCDAB89u)
#define X16_2(v) xor16p((v), 0x54761032u, 0xDCFE98BAu)
#define X16_4(v) xor16p((v), 0x32107654u, 0xBA98FEDCu)
#define X16_8(v) xor16p((v), 0xFEDCBA98u, 0x76543210u)
#else
#define X16_1(v) __shfl_xor((v), 1, 32)
#define X16_2(v) __shfl_xor((v), 2, 32)
#define X16_4(v) __shfl_xor((v), 4, 32)
#define X16_8(v) __shfl_xor((v), 8, 32)
#endif

// ---- CDNA5 async global->LDS copy (ASYNCcnt path), reg-pipeline fallback ----
#if __has_builtin(__builtin_amdgcn_global_load_async_to_lds_b128) && \
    __has_builtin(__builtin_amdgcn_s_wait_asynccnt)
#define ASYNC_LDS 1
#endif

// ---- Non-temporal (TH=NT) streaming access for single-use HBM traffic ----
#if __has_builtin(__builtin_nontemporal_load) && __has_builtin(__builtin_nontemporal_store)
#define NT_LOAD(p)     __builtin_nontemporal_load(p)
#define NT_STORE(v, p) __builtin_nontemporal_store((v), (p))
#else
#define NT_LOAD(p)     (*(p))
#define NT_STORE(v, p) (*(p) = (v))
#endif

// ---------------------------------------------------------------------------
// k0: transpose+convert Wv to f16 so B-fragments load as contiguous 16B runs.
// ---------------------------------------------------------------------------
__global__ __launch_bounds__(256) void k0_prep(const float* __restrict__ Wv,
                                               _Float16* __restrict__ wvh) {
    int i = blockIdx.x * 256 + threadIdx.x;   // 64 blocks * 256 = 16384
    int n = i >> 7, k = i & 127;
    wvh[i] = (_Float16)Wv[k * 128 + n];
}

// ---------------------------------------------------------------------------
// k1: per-batch q path: LN1 -> W1 -> gelu -> q_atten ; LNq -> Wq -> qh ;
//     qkb[b][h] = dot(qh_h, bk_h) ;  U[b][h][d] = sum_n Wk[d][n]*qh_h[n]
// ---------------------------------------------------------------------------
__global__ __launch_bounds__(256) void k1_qpath(
    const float* __restrict__ q_feats,
    const float* __restrict__ ln1_g, const float* __restrict__ ln1_b,
    const float* __restrict__ W1,    const float* __restrict__ b1,
    const float* __restrict__ lnq_g, const float* __restrict__ lnq_b,
    const float* __restrict__ Wq,    const float* __restrict__ bq,
    const float* __restrict__ Wk,    const float* __restrict__ bk,
    float* __restrict__ q_atten, float* __restrict__ U, float* __restrict__ qkb)
{
    const int b = blockIdx.x, t = threadIdx.x;
    __shared__ float xs[256];
    __shared__ float red[256];
    __shared__ float ya[128];
    __shared__ float qhs[128];

    // LayerNorm over 256
    float x = q_feats[b * 256 + t];
    red[t] = x; __syncthreads();
    for (int s = 128; s > 0; s >>= 1) { if (t < s) red[t] += red[t + s]; __syncthreads(); }
    float mu = red[0] * (1.f / 256.f); __syncthreads();
    float d0 = x - mu;
    red[t] = d0 * d0; __syncthreads();
    for (int s = 128; s > 0; s >>= 1) { if (t < s) red[t] += red[t + s]; __syncthreads(); }
    float inv = rsqrtf(red[0] * (1.f / 256.f) + 1e-6f); __syncthreads();
    xs[t] = d0 * inv * ln1_g[t] + ln1_b[t];
    __syncthreads();

    // x @ W1 + b1 -> gelu
    {
        int n = t & 127, h = t >> 7;
        float acc = 0.f;
        const float* wc = W1 + n;
        for (int d = h * 128; d < h * 128 + 128; ++d) acc = fmaf(xs[d], wc[d * 128], acc);
        red[t] = acc;
    }
    __syncthreads();
    if (t < 128) {
        float o = gelu_exact(red[t] + red[t + 128] + b1[t]);
        ya[t] = o;
        q_atten[b * 128 + t] = o;
    }
    __syncthreads();

    // LayerNorm over 128
    red[t] = (t < 128) ? ya[t] : 0.f; __syncthreads();
    for (int s = 128; s > 0; s >>= 1) { if (t < s) red[t] += red[t + s]; __syncthreads(); }
    float mu2 = red[0] * (1.f / 128.f); __syncthreads();
    float d2 = (t < 128) ? (ya[t] - mu2) : 0.f;
    red[t] = d2 * d2; __syncthreads();
    for (int s = 128; s > 0; s >>= 1) { if (t < s) red[t] += red[t + s]; __syncthreads(); }
    float inv2 = rsqrtf(red[0] * (1.f / 128.f) + 1e-6f); __syncthreads();
    if (t < 128) xs[t] = d2 * inv2 * lnq_g[t] + lnq_b[t];
    __syncthreads();

    // qh = q_in @ Wq + bq
    {
        int n = t & 127, h = t >> 7;
        float acc = 0.f;
        const float* wc = Wq + n;
        for (int d = h * 64; d < h * 64 + 64; ++d) acc = fmaf(xs[d], wc[d * 128], acc);
        red[t] = acc;
    }
    __syncthreads();
    if (t < 128) qhs[t] = red[t] + red[t + 128] + bq[t];
    __syncthreads();

    // qkb[h] = dot(qh_h, bk_h)  (deterministic serial tail)
    red[t] = (t < 128) ? qhs[t] * bk[t] : 0.f;
    __syncthreads();
    if (t < 2) {
        float s = 0.f;
        for (int i = 0; i < 64; ++i) s += red[t * 64 + i];
        qkb[b * 2 + t] = s;
    }
    // U[h][d] = sum_{n in head h} Wk[d][n] * qh[n]
    {
        int d = t & 127, h = t >> 7;
        const float* wr = Wk + d * 128 + h * 64;
        const float* qv = qhs + h * 64;
        float acc = 0.f;
        for (int n = 0; n < 64; ++n) acc = fmaf(wr[n], qv[n], acc);
        U[b * 256 + h * 128 + d] = acc;
    }
}

// ---------------------------------------------------------------------------
// k2: fused masked attention over k_feats. One block per batch; 8 waves;
// wave w owns V-output columns [16w,16w+16). Double-buffered tile pipeline:
// tile i+1 streams into LDS (async-to-LDS if available) while tile i runs
// the fused LN+score pass, the Wv WMMA GEMM, and online masked softmax.
// ---------------------------------------------------------------------------
__global__ __launch_bounds__(256) void k2_attn(
    const float*    __restrict__ k_feats,
    const int*      __restrict__ msk,
    const float*    __restrict__ lnk_g, const float* __restrict__ lnk_b,
    const _Float16* __restrict__ wvh,
    const float*    __restrict__ U,     const float* __restrict__ qkb,
    float*          __restrict__ ctx)
{
    const int b    = blockIdx.x;
    const int t    = threadIdx.x;
    const int lane = t & 31;
    const int w    = t >> 5;        // wave 0..7
    const int hw   = w >> 2;        // head for this wave's columns
    const int l16  = lane & 15;
    const int hi   = lane >> 4;     // lane half (selects M rows / K groups)
    const int n0   = w * 16;
    const int r    = t >> 4;        // tile row handled by this thread
    const int c0   = (t & 15) * 8;  // first of 8 columns handled

    __shared__ float    tileF[2][16][128];   // double-buffered key tile
    __shared__ _Float16 kvH[16][128];
    __shared__ float    gBuf[128], bBuf[128];
    __shared__ float    gu[2][128];          // lnk_g * u_h
    __shared__ float    scoreBuf[2][16];     // masked scores
    __shared__ float    pBuf[2][16];
    __shared__ float    mlsc[6];             // M0 M1 L0 L1 sc0 sc1
    __shared__ float    hcs[4];              // G0 G1 Cb0 Cb1

#if ASYNC_LDS
#define AS1V(p) ((__attribute__((address_space(1))) i32x4*)(p))
#define AS3V(p) ((__attribute__((address_space(3))) i32x4*)(p))
#define TILE_ISSUE(buf, kk)                                                          \
    do {                                                                             \
        const float* _g = k_feats + (((long)b * LK_ + (kk) + r) << 7) + c0;          \
        __builtin_amdgcn_global_load_async_to_lds_b128(                              \
            AS1V(_g), AS3V(&tileF[buf][r][c0]), 0, 0);                               \
        __builtin_amdgcn_global_load_async_to_lds_b128(                              \
            AS1V(_g + 4), AS3V(&tileF[buf][r][c0 + 4]), 0, 0);                       \
    } while (0)
#define TILE_COMMIT(buf) __builtin_amdgcn_s_wait_asynccnt(0)
#else
    f32x4 nb0{}, nb1{};
#define TILE_ISSUE(buf, kk)                                                          \
    do {                                                                             \
        const f32x4* _g =                                                            \
            (const f32x4*)(k_feats + (((long)b * LK_ + (kk) + r) << 7) + c0);        \
        nb0 = _g[0]; nb1 = _g[1];                                                    \
    } while (0)
#define TILE_COMMIT(buf)                                                             \
    do {                                                                             \
        *(f32x4*)&tileF[buf][r][c0]     = nb0;                                       \
        *(f32x4*)&tileF[buf][r][c0 + 4] = nb1;                                       \
    } while (0)
#endif

    if (t < 128) { gBuf[t] = lnk_g[t]; bBuf[t] = lnk_b[t]; }
    if (t < 2)   { mlsc[t] = -__builtin_inff(); mlsc[2 + t] = 0.f; }
    __syncthreads();
    if (t < 128) {
        float u0 = U[b * 256 + t], u1 = U[b * 256 + 128 + t];
        float g0 = gBuf[t] * u0, g1 = gBuf[t] * u1;
        gu[0][t] = g0; gu[1][t] = g1;
        tileF[0][0][t] = g0;            tileF[0][1][t] = g1;
        tileF[0][2][t] = bBuf[t] * u0;  tileF[0][3][t] = bBuf[t] * u1;
    }
    __syncthreads();
    if (t < 4) {                       // deterministic reductions for G/Cb
        float s = 0.f;
        for (int i = 0; i < 128; ++i) s += tileF[0][t][i];
        hcs[t] = s;
    }
    // Wv B-fragments resident in VGPRs (contiguous k = 16*hi + s, same order as A)
    v16h bvf[4];
#pragma unroll
    for (int c = 0; c < 4; ++c) {
        const _Float16* vp = wvh + (n0 + l16) * 128 + c * 32 + hi * 16;
        v8h vlo = *(const v8h*)vp, vhi2 = *(const v8h*)(vp + 8);
        bvf[c] = __builtin_shufflevector(vlo, vhi2, 0,1,2,3,4,5,6,7,8,9,10,11,12,13,14,15);
    }
    const float qkbh0 = qkb[b * 2 + 0], qkbh1 = qkb[b * 2 + 1];
    float accv = 0.f;
    __syncthreads();                    // scratch reads done; safe to fill tile 0

    TILE_ISSUE(0, 0);
    TILE_COMMIT(0);
    __syncthreads();                    // publish tile 0

    for (int k0 = 0; k0 < LK_; k0 += 16) {
        const int  cur = (k0 >> 4) & 1, nxt = cur ^ 1;
        const bool have_nxt = (k0 + 16) < LK_;
        if (have_nxt) TILE_ISSUE(nxt, k0 + 16);   // stream next tile under compute

        // fused LayerNorm + score pass (16 threads per row, permlane reduce);
        // row leaders also fetch the mask here so its latency hides under the
        // reduce chain and the softmax phase has no VMEM at all.
        {
            int mv = 0;
            if ((t & 15) == 0) mv = msk[b * LK_ + k0 + r];
            float s = 0.f, ss = 0.f, e0 = 0.f, e1 = 0.f, xv[8];
#pragma unroll
            for (int i = 0; i < 8; ++i) {
                float v = tileF[cur][r][c0 + i]; xv[i] = v;
                s += v; ss = fmaf(v, v, ss);
                e0 = fmaf(v, gu[0][c0 + i], e0);
                e1 = fmaf(v, gu[1][c0 + i], e1);
            }
            s += X16_1(s); ss += X16_1(ss); e0 += X16_1(e0); e1 += X16_1(e1);
            s += X16_2(s); ss += X16_2(ss); e0 += X16_2(e0); e1 += X16_2(e1);
            s += X16_4(s); ss += X16_4(ss); e0 += X16_4(e0); e1 += X16_4(e1);
            s += X16_8(s); ss += X16_8(ss); e0 += X16_8(e0); e1 += X16_8(e1);
            float mu  = s * (1.f / 128.f);
            float inv = rsqrtf(ss * (1.f / 128.f) - mu * mu + 1e-6f);
#pragma unroll
            for (int i = 0; i < 8; ++i) {
                int cc = c0 + i;
                kvH[r][cc] = (_Float16)((xv[i] - mu) * inv * gBuf[cc] + bBuf[cc]);
            }
            if ((t & 15) == 0) {
                float s0 = 0.125f * (fmaf(inv, e0 - mu * hcs[0], hcs[2]) + qkbh0);
                float s1 = 0.125f * (fmaf(inv, e1 - mu * hcs[1], hcs[3]) + qkbh1);
                scoreBuf[0][r] = mv ? s0 : -1e9f;
                scoreBuf[1][r] = mv ? s1 : -1e9f;
            }
        }
        __syncthreads();                                       // (2)
        // preload all A-fragments (8 pipelined ds_load_b128), then 4 WMMAs
        v16h af[4];
#pragma unroll
        for (int c = 0; c < 4; ++c) {
            const _Float16* ap = &kvH[l16][c * 32 + hi * 16];
            v8h alo  = *(const v8h*)ap;
            v8h ahi2 = *(const v8h*)(ap + 8);
            af[c] = __builtin_shufflevector(alo, ahi2, 0,1,2,3,4,5,6,7,8,9,10,11,12,13,14,15);
        }
        v8f cvv = {0,0,0,0,0,0,0,0};
#pragma unroll
        for (int c = 0; c < 4; ++c)
            cvv = __builtin_amdgcn_wmma_f32_16x16x32_f16(false, af[c], false, bvf[c],
                                                         (short)0, cvv, false, false);
        // wave 0 lanes 0..15: online softmax bookkeeping (LDS + VALU only)
        if (t < 16) {
            float s0 = scoreBuf[0][t];
            float s1 = scoreBuf[1][t];
            float m0 = s0, m1 = s1;
            m0 = fmaxf(m0, X16_1(m0)); m1 = fmaxf(m1, X16_1(m1));
            m0 = fmaxf(m0, X16_2(m0)); m1 = fmaxf(m1, X16_2(m1));
            m0 = fmaxf(m0, X16_4(m0)); m1 = fmaxf(m1, X16_4(m1));
            m0 = fmaxf(m0, X16_8(m0)); m1 = fmaxf(m1, X16_8(m1));
            float M0o = mlsc[0], M1o = mlsc[1];
            float M0n = fmaxf(M0o, m0), M1n = fmaxf(M1o, m1);
            float p0 = __expf(s0 - M0n), p1 = __expf(s1 - M1n);
            pBuf[0][t] = p0; pBuf[1][t] = p1;
            float l0 = p0, l1 = p1;
            l0 += X16_1(l0); l1 += X16_1(l1);
            l0 += X16_2(l0); l1 += X16_2(l1);
            l0 += X16_4(l0); l1 += X16_4(l1);
            l0 += X16_8(l0); l1 += X16_8(l1);
            if (t == 0) {
                float sc0 = __expf(M0o - M0n), sc1 = __expf(M1o - M1n);
                mlsc[4] = sc0; mlsc[5] = sc1;
                mlsc[2] = mlsc[2] * sc0 + l0;
                mlsc[3] = mlsc[3] * sc1 + l1;
                mlsc[0] = M0n; mlsc[1] = M1n;
            }
        }
        __syncthreads();                                       // (3)
        // ctx accumulation (rescale + add probability-weighted V)
        {
            float sc = mlsc[4 + hw];
            accv *= sc;
#pragma unroll
            for (int j = 0; j < 8; ++j) accv = fmaf(pBuf[hw][hi * 8 + j], cvv[j], accv);
        }
        if (have_nxt) TILE_COMMIT(nxt);
        __syncthreads();                                       // (1) publish next
    }
    // combine lane halves and normalize
    float other = __shfl_xor(accv, 16, 32);
    if (hi == 0) {
        float L = mlsc[2 + hw];
        ctx[(long)b * 128 + n0 + l16] = (accv + other) / L;
    }
#undef TILE_ISSUE
#undef TILE_COMMIT
}

// ---------------------------------------------------------------------------
// k3: per-batch tail: attn_out = gelu((ctx+bv)@Wo+bo); q_atten += attn_out;
//     scaled = q_atten*scale_w ; q_out = q_feats + gelu(LN2(q_atten)@W2+b2)
// ---------------------------------------------------------------------------
__global__ __launch_bounds__(256) void k3_qout(
    const float* __restrict__ q_feats, const float* __restrict__ ctx,
    const float* __restrict__ bv,
    const float* __restrict__ Wo, const float* __restrict__ bo,
    const float* __restrict__ q_atten, const float* __restrict__ scale_w,
    const float* __restrict__ ln2_g, const float* __restrict__ ln2_b,
    const float* __restrict__ W2, const float* __restrict__ b2,
    float* __restrict__ scaled, float* __restrict__ q_out)
{
    const int b = blockIdx.x, t = threadIdx.x;
    __shared__ float cf[128], red[256], ya[128], yn[128];
    if (t < 128) cf[t] = ctx[b * 128 + t] + bv[t];
    __syncthreads();
    {
        int n = t & 127, h = t >> 7;
        float acc = 0.f;
        const float* wc = Wo + n;
        for (int d = h * 64; d < h * 64 + 64; ++d) acc = fmaf(cf[d], wc[d * 128], acc);
        red[t] = acc;
    }
    __syncthreads();
    if (t < 128) {
        float att = gelu_exact(red[t] + red[t + 128] + bo[t]);
        float qa2 = q_atten[b * 128 + t] + att;
        ya[t] = qa2;
        scaled[b * 128 + t] = qa2 * scale_w[t];
    }
    __syncthreads();
    red[t] = (t < 128) ? ya[t] : 0.f; __syncthreads();
    for (int s = 128; s > 0; s >>= 1) { if (t < s) red[t] += red[t + s]; __syncthreads(); }
    float mu = red[0] * (1.f / 128.f); __syncthreads();
    float dd = (t < 128) ? (ya[t] - mu) : 0.f;
    red[t] = dd * dd; __syncthreads();
    for (int s = 128; s > 0; s >>= 1) { if (t < s) red[t] += red[t + s]; __syncthreads(); }
    float inv = rsqrtf(red[0] * (1.f / 128.f) + 1e-6f); __syncthreads();
    if (t < 128) yn[t] = dd * inv * ln2_g[t] + ln2_b[t];
    __syncthreads();
    {
        float acc = 0.f;
        const float* wc = W2 + t;           // W2 [128][256]
        for (int d = 0; d < 128; ++d) acc = fmaf(yn[d], wc[d * 256], acc);
        q_out[b * 256 + t] = q_feats[b * 256 + t] + gelu_exact(acc + b2[t]);
    }
}

// ---------------------------------------------------------------------------
// k4: k_out = k_feats + broadcast(scaled[j]).  1 GB of strictly single-use
// streaming traffic -> non-temporal (TH=NT) load+store so it doesn't evict
// the small, heavily reused scaled[] table from L2; scaled stays RT-cached.
// ---------------------------------------------------------------------------
__global__ __launch_bounds__(256) void k4_kout(
    const float* __restrict__ kf, const float* __restrict__ scaled,
    float* __restrict__ kout)
{
    long i  = (long)blockIdx.x * 256 + threadIdx.x;  // float4 index, 2^25 total
    long row = i >> 5;                               // 32 float4 per row
    int  j   = (int)(row & (LK_ - 1));               // key index -> q_atten[j]
    int  c4  = (int)(i & 31);
    f32x4 a = NT_LOAD((const f32x4*)kf + i);
    f32x4 s = *((const f32x4*)scaled + j * 32 + c4);
    NT_STORE(a + s, (f32x4*)kout + i);
}

// ---------------------------------------------------------------------------
extern "C" void kernel_launch(void* const* d_in, const int* in_sizes, int n_in,
                              void* d_out, int out_size, void* d_ws, size_t ws_size,
                              hipStream_t stream)
{
    (void)in_sizes; (void)n_in; (void)out_size; (void)ws_size;
    const float* q_feats = (const float*)d_in[0];
    const float* k_feats = (const float*)d_in[1];
    const int*   msk     = (const int*)  d_in[2];
    const float* ln1_g = (const float*)d_in[3];
    const float* ln1_b = (const float*)d_in[4];
    const float* W1    = (const float*)d_in[5];
    const float* b1    = (const float*)d_in[6];
    const float* lnq_g = (const float*)d_in[7];
    const float* lnq_b = (const float*)d_in[8];
    const float* lnk_g = (const float*)d_in[9];
    const float* lnk_b = (const float*)d_in[10];
    const float* Wq = (const float*)d_in[11];
    const float* bq = (const float*)d_in[12];
    const float* Wk = (const float*)d_in[13];
    const float* bk = (const float*)d_in[14];
    const float* Wv = (const float*)d_in[15];
    const float* bv = (const float*)d_in[16];
    const float* Wo = (const float*)d_in[17];
    const float* bo = (const float*)d_in[18];
    const float* scale_w = (const float*)d_in[19];
    const float* ln2_g = (const float*)d_in[20];
    const float* ln2_b = (const float*)d_in[21];
    const float* W2 = (const float*)d_in[22];
    const float* b2 = (const float*)d_in[23];

    char* ws = (char*)d_ws;
    float* q_atten = (float*)ws; ws += (size_t)B_ * 128 * 4;
    float* ctxb    = (float*)ws; ws += (size_t)B_ * 128 * 4;
    float* scaled  = (float*)ws; ws += (size_t)B_ * 128 * 4;
    float* qkb     = (float*)ws; ws += (size_t)B_ * 2 * 4;
    float* U       = (float*)ws; ws += (size_t)B_ * 256 * 4;
    _Float16* wvh  = (_Float16*)ws; ws += (size_t)128 * 128 * 2;

    float* q_out = (float*)d_out;
    float* k_out = (float*)d_out + (size_t)B_ * LQD_;

    k0_prep <<<64,     256, 0, stream>>>(Wv, wvh);
    k1_qpath<<<B_,     256, 0, stream>>>(q_feats, ln1_g, ln1_b, W1, b1,
                                         lnq_g, lnq_b, Wq, bq, Wk, bk,
                                         q_atten, U, qkb);
    k2_attn <<<B_,     256, 0, stream>>>(k_feats, msk, lnk_g, lnk_b,
                                         wvh, U, qkb, ctxb);
    k3_qout <<<B_,     256, 0, stream>>>(q_feats, ctxb, bv, Wo, bo,
                                         q_atten, scale_w, ln2_g, ln2_b,
                                         W2, b2, scaled, q_out);
    k4_kout <<<131072, 256, 0, stream>>>(k_feats, scaled, (float*)d_out + (size_t)B_ * LQD_);
    (void)k_out;
}
